// GraphEncoder_23210003268200
// MI455X (gfx1250) — matrix operand order repaired and checked by reading, not compiled, not run on hardware
//
#include <hip/hip_runtime.h>

#define N_NODES 100000
#define E_EDGES 1600000
#define DIN 128
#define DHID 256
#define DOUT 128

typedef __attribute__((ext_vector_type(16))) __bf16 v16bf;
typedef __attribute__((ext_vector_type(8)))  float  v8f;

// Explicit CDNA5 hardware float atomic: single L2-side fp32 add, device scope,
// no CAS retry loop. Tracked by STOREcnt; S_ENDPGM's implicit wait-idle plus
// stream ordering makes results visible to the following kernel.
__device__ __forceinline__ void gatomic_fadd(float* p, float v) {
  asm volatile("global_atomic_add_f32 %0, %1, off scope:SCOPE_DEV"
               :: "v"(p), "v"(v) : "memory");
}

// CDNA5 async copy engine: 16 bytes global -> LDS per lane, no VGPR data path,
// tracked by ASYNCcnt (ISA 10.x / 15.18.3 GLOBAL_LOAD_ASYNC_TO_LDS_B128).
__device__ __forceinline__ void async_g2lds_b128(__bf16* lds_dst, const void* g) {
  auto ldsp = (__attribute__((address_space(3))) __bf16*)lds_dst;
  asm volatile("global_load_async_to_lds_b128 %0, %1, off"
               :: "v"(ldsp), "v"(g) : "memory");
}
__device__ __forceinline__ void wait_async0() {
  asm volatile("s_wait_asynccnt 0x0" ::: "memory");
}

// ---------------------------------------------------------------------------
// Pack fp32 weights into bf16, pre-swizzled into WMMA B-fragment order:
//   Bpack[kstep][ntile][lane][16 bf16], element e of lane L:
//     K = kstep*32 + (L>>4)*16 + e,  N = ntile*16 + (L&15)
// Layer1: combined K=256 ([Wl1;Wr1], 256x256), 16 ntiles, 8 ksteps -> 65536 el
// Layer2: combined K=512 ([Wl2;Wr2], 512x128),  8 ntiles, 16 ksteps -> 65536 el
// ---------------------------------------------------------------------------
__global__ __launch_bounds__(256) void pack_weights(
    const float* __restrict__ Wl1, const float* __restrict__ Wr1,
    const float* __restrict__ Wl2, const float* __restrict__ Wr2,
    __bf16* __restrict__ bp1, __bf16* __restrict__ bp2)
{
  int idx = blockIdx.x * 256 + threadIdx.x;
  if (idx < 65536) {
    int e    = idx & 15;
    int lane = (idx >> 4) & 31;
    int nt   = (idx >> 9) & 15;
    int ks   = idx >> 13;                 // 0..7
    int K    = ks * 32 + (lane >> 4) * 16 + e;
    int n    = nt * 16 + (lane & 15);
    float v  = (K < DIN) ? Wl1[K * DHID + n] : Wr1[(K - DIN) * DHID + n];
    bp1[idx] = (__bf16)v;
  } else {
    int i2   = idx - 65536;
    int e    = i2 & 15;
    int lane = (i2 >> 4) & 31;
    int nt   = (i2 >> 9) & 7;
    int ks   = i2 >> 12;                  // 0..15
    int K    = ks * 32 + (lane >> 4) * 16 + e;
    int n    = nt * 16 + (lane & 15);
    float v  = (K < DHID) ? Wl2[K * DOUT + n] : Wr2[(K - DHID) * DOUT + n];
    bp2[i2]  = (__bf16)v;
  }
}

// ---------------------------------------------------------------------------
// scatter1: agg1[row[e]][0:128] += x[col[e]][0:128], one wave32 per edge,
// grid-stride loop with prefetch of the next edge's source row.
// x and agg1 are both L2-resident (51 MB each vs 192 MB L2).
// ---------------------------------------------------------------------------
__global__ __launch_bounds__(256) void scatter1(
    const float* __restrict__ x, const int* __restrict__ row,
    const int* __restrict__ col, float* __restrict__ agg)
{
  const int wave = threadIdx.x >> 5;
  const int lane = threadIdx.x & 31;
  const long stride = (long)gridDim.x * 8;
  for (long e = (long)blockIdx.x * 8 + wave; e < E_EDGES; e += stride) {
    long en = e + stride;
    if (en < E_EDGES) {  // prefetch next source row into L0/L2
      int cn = col[en];
      __builtin_prefetch(x + (long)cn * DIN + lane * 4, 0, 0);
    }
    int r = row[e], c = col[e];
    const float4 v = *(const float4*)(x + (long)c * DIN + lane * 4);
    float* dst = agg + (long)r * DIN + lane * 4;
    gatomic_fadd(dst + 0, v.x);
    gatomic_fadd(dst + 1, v.y);
    gatomic_fadd(dst + 2, v.z);
    gatomic_fadd(dst + 3, v.w);
  }
}

// scatter2: agg2[row[e]][0:256] += h1[col[e]][0:256]
__global__ __launch_bounds__(256) void scatter2(
    const float* __restrict__ h1, const int* __restrict__ row,
    const int* __restrict__ col, float* __restrict__ agg)
{
  const int wave = threadIdx.x >> 5;
  const int lane = threadIdx.x & 31;
  const long stride = (long)gridDim.x * 8;
  for (long e = (long)blockIdx.x * 8 + wave; e < E_EDGES; e += stride) {
    long en = e + stride;
    if (en < E_EDGES) {
      int cn = col[en];
      __builtin_prefetch(h1 + (long)cn * DHID + lane * 4, 0, 0);
      __builtin_prefetch(h1 + (long)cn * DHID + 128 + lane * 4, 0, 0);
    }
    int r = row[e], c = col[e];
    const float* src = h1 + (long)c * DHID;
    float* dst = agg + (long)r * DHID;
    #pragma unroll
    for (int half = 0; half < 2; ++half) {
      int off = half * 128 + lane * 4;
      const float4 v = *(const float4*)(src + off);
      gatomic_fadd(dst + off + 0, v.x);
      gatomic_fadd(dst + off + 1, v.y);
      gatomic_fadd(dst + off + 2, v.z);
      gatomic_fadd(dst + off + 3, v.w);
    }
  }
}

// ---------------------------------------------------------------------------
// gemm1: h1 = relu([agg1 | x] @ [Wl1;Wr1]_bf16 + bl1)   (K=256, ncols=256)
// Full packed B (128 KB) is staged once per block into LDS (320 KB/WGP on
// CDNA5) via the async global->LDS engine; block then loops over 10 row
// tiles, restaging only the 8 KB A tile.
// A-fragment LDS order: As[kstep][lane][16 bf16]; lane L: row = L&15,
// element e holds K = kstep*32 + 16*(L>>4) + e.
// ---------------------------------------------------------------------------
#define TILES_PER_BLOCK 10   // 6250 tiles = 625 blocks * 10

__global__ __launch_bounds__(256) void gemm1(
    const float* __restrict__ agg1, const float* __restrict__ x,
    const __bf16* __restrict__ bp1, const float* __restrict__ bias,
    float* __restrict__ h1)
{
  __shared__ __attribute__((aligned(32))) __bf16 Bs[65536];     // 128 KB
  __shared__ __attribute__((aligned(32))) __bf16 As[8 * 32 * 16]; // 8 KB
  const int t = threadIdx.x;

  // async-stage whole packed B into LDS once: 8192 x b128, no VGPR data path
  for (int i = t; i < 8192; i += 256)
    async_g2lds_b128(&Bs[i * 8], (const char*)bp1 + i * 16);
  wait_async0();
  __syncthreads();

  const int wave = t >> 5, lane = t & 31;
  const int nt0 = wave * 2, nt1 = wave * 2 + 1;
  const int r = lane & 15, kh = lane >> 4;

  for (int tt = 0; tt < TILES_PER_BLOCK; ++tt) {
    const long node0 = ((long)blockIdx.x * TILES_PER_BLOCK + tt) * 16;
    {  // stage A: thread t fills fragment slot (kstep = t>>5, lane = t&31)
      int ks = t >> 5;
      long nrow = node0 + r;
      #pragma unroll
      for (int v = 0; v < 8; ++v) {
        int kk = 2 * v + 16 * kh;
        int k = ks * 32 + kk;                 // even; pair never straddles 128
        float2 f = (k < DIN) ? *(const float2*)(agg1 + nrow * DIN + k)
                             : *(const float2*)(x + nrow * DIN + (k - DIN));
        As[t * 16 + 2 * v]     = (__bf16)f.x;
        As[t * 16 + 2 * v + 1] = (__bf16)f.y;
      }
    }
    __syncthreads();

    v8f c0 = {}, c1 = {};
    #pragma unroll
    for (int ks = 0; ks < 8; ++ks) {
      v16bf a  = *reinterpret_cast<const v16bf*>(&As[(ks * 32 + lane) * 16]);
      v16bf b0 = *reinterpret_cast<const v16bf*>(&Bs[((ks * 16 + nt0) * 32 + lane) * 16]);
      v16bf b1 = *reinterpret_cast<const v16bf*>(&Bs[((ks * 16 + nt1) * 32 + lane) * 16]);
      c0 = __builtin_amdgcn_wmma_f32_16x16x32_bf16(false, a, false, b0, (short)0, c0, false, false);
      c1 = __builtin_amdgcn_wmma_f32_16x16x32_bf16(false, a, false, b1, (short)0, c1, false, false);
    }
    __syncthreads();  // all waves done with As before next restage

    const int n = lane & 15;
    #pragma unroll
    for (int i = 0; i < 8; ++i) {
      long nrow = node0 + i + 8 * kh;
      int col0 = nt0 * 16 + n;
      int col1 = nt1 * 16 + n;
      float v0 = c0[i] + bias[col0];
      float v1 = c1[i] + bias[col1];
      h1[nrow * DHID + col0] = v0 > 0.f ? v0 : 0.f;
      h1[nrow * DHID + col1] = v1 > 0.f ? v1 : 0.f;
    }
  }
}

// ---------------------------------------------------------------------------
// gemm2: out = [agg2 | h1] @ [Wl2;Wr2]_bf16 + bl2   (K=512, ncols=128)
// ---------------------------------------------------------------------------
__global__ __launch_bounds__(256) void gemm2(
    const float* __restrict__ agg2, const float* __restrict__ h1,
    const __bf16* __restrict__ bp2, const float* __restrict__ bias,
    float* __restrict__ out)
{
  __shared__ __attribute__((aligned(32))) __bf16 Bs[65536];       // 128 KB
  __shared__ __attribute__((aligned(32))) __bf16 As[16 * 32 * 16]; // 16 KB
  const int t = threadIdx.x;

  for (int i = t; i < 8192; i += 256)
    async_g2lds_b128(&Bs[i * 8], (const char*)bp2 + i * 16);
  wait_async0();
  __syncthreads();

  const int wave = t >> 5, lane = t & 31;  // wave = column tile (8 tiles)
  const int kh = lane >> 4;

  for (int tt = 0; tt < TILES_PER_BLOCK; ++tt) {
    const long node0 = ((long)blockIdx.x * TILES_PER_BLOCK + tt) * 16;
    #pragma unroll
    for (int half = 0; half < 2; ++half) {  // 512 fragment slots, 256 threads
      int s = t + half * 256;
      int ks = s >> 5, sl = s & 31;
      long nrow = node0 + (sl & 15);
      int skh = sl >> 4;
      #pragma unroll
      for (int v = 0; v < 8; ++v) {
        int kk = 2 * v + 16 * skh;
        int k = ks * 32 + kk;
        float2 f = (k < DHID) ? *(const float2*)(agg2 + nrow * DHID + k)
                              : *(const float2*)(h1 + nrow * DHID + (k - DHID));
        As[s * 16 + 2 * v]     = (__bf16)f.x;
        As[s * 16 + 2 * v + 1] = (__bf16)f.y;
      }
    }
    __syncthreads();

    v8f c = {};
    #pragma unroll
    for (int ks = 0; ks < 16; ++ks) {
      v16bf a = *reinterpret_cast<const v16bf*>(&As[(ks * 32 + lane) * 16]);
      v16bf b = *reinterpret_cast<const v16bf*>(&Bs[((ks * 8 + wave) * 32 + lane) * 16]);
      c = __builtin_amdgcn_wmma_f32_16x16x32_bf16(false, a, false, b, (short)0, c, false, false);
    }
    __syncthreads();

    const int n = lane & 15;
    #pragma unroll
    for (int i = 0; i < 8; ++i) {
      long nrow = node0 + i + 8 * kh;
      int col = wave * 16 + n;
      out[nrow * DOUT + col] = c[i] + bias[col];
    }
  }
}

// ---------------------------------------------------------------------------
extern "C" void kernel_launch(void* const* d_in, const int* in_sizes, int n_in,
                              void* d_out, int out_size, void* d_ws, size_t ws_size,
                              hipStream_t stream) {
  (void)in_sizes; (void)n_in; (void)out_size; (void)ws_size;
  const float* x   = (const float*)d_in[0];
  const int*   row = (const int*)d_in[1];
  const int*   col = (const int*)d_in[2];
  const float* Wl1 = (const float*)d_in[3];
  const float* bl1 = (const float*)d_in[4];
  const float* Wr1 = (const float*)d_in[5];
  const float* Wl2 = (const float*)d_in[6];
  const float* bl2 = (const float*)d_in[7];
  const float* Wr2 = (const float*)d_in[8];
  float* outp = (float*)d_out;

  // workspace layout (fp32 elements): agg1 | h1 | agg2 | bp1(bf16) | bp2(bf16)
  float* agg1 = (float*)d_ws;                       // N*DIN   = 12.8 M
  float* h1   = agg1 + (size_t)N_NODES * DIN;       // N*DHID  = 25.6 M
  float* agg2 = h1   + (size_t)N_NODES * DHID;      // N*DHID  = 25.6 M
  __bf16* bp1 = (__bf16*)(agg2 + (size_t)N_NODES * DHID);  // 65536 bf16
  __bf16* bp2 = bp1 + 65536;                               // 65536 bf16

  const int scatter_blocks = 8192;      // grid-stride, ~24 edges per wave
  const int gemm_blocks    = 625;       // 625 * 10 tiles * 16 rows = 100000

  pack_weights<<<512, 256, 0, stream>>>(Wl1, Wr1, Wl2, Wr2, bp1, bp2);

  hipMemsetAsync(agg1, 0, (size_t)N_NODES * DIN * sizeof(float), stream);
  scatter1<<<scatter_blocks, 256, 0, stream>>>(x, row, col, agg1);
  gemm1<<<gemm_blocks, 256, 0, stream>>>(agg1, x, bp1, bl1, h1);

  hipMemsetAsync(agg2, 0, (size_t)N_NODES * DHID * sizeof(float), stream);
  scatter2<<<scatter_blocks, 256, 0, stream>>>(h1, row, col, agg2);
  gemm2<<<gemm_blocks, 256, 0, stream>>>(agg2, h1, bp2, bl2, outp);
}
